// ProbSparseAttention_6262062317957
// MI455X (gfx1250) — matrix operand order repaired and checked by reading, not compiled, not run on hardware
//
#include <hip/hip_runtime.h>
#include <hip/hip_bf16.h>
#include <math.h>

// ---------------------------------------------------------------------------
// ProbSparse attention for MI455X (gfx1250, wave32, WMMA)
// B=8, L=4096, D=512, u=41
// ---------------------------------------------------------------------------

#define BB 8
#define LL 4096
#define DD 512
#define UU 41
#define QROWS 128            // query rows per workgroup in the M kernel
#define KTILE 16             // keys per staged tile
#define NKT (LL / KTILE)     // 256 key tiles

typedef __attribute__((ext_vector_type(16))) _Float16 v16h;
typedef __attribute__((ext_vector_type(8)))  float    v8f;

static constexpr float SCALE = 0.04419417382415922f;   // 1/sqrt(512)

// ---- workspace layout (bytes) ---------------------------------------------
static constexpr size_t K_HI_OFF   = 0;                                   // 32 MB f16
static constexpr size_t K_LO_OFF   = K_HI_OFF + (size_t)BB*LL*DD*2;       // 32 MB f16
static constexpr size_t M_OFF      = K_LO_OFF + (size_t)BB*LL*DD*2;       // 128 KB f32
static constexpr size_t TOPK_OFF   = M_OFF    + (size_t)BB*LL*4;          // 1.5 KB int
static constexpr size_t VPART_OFF  = TOPK_OFF + (size_t)BB*48*4;          // 256 KB f32
static constexpr size_t VMEAN_OFF  = VPART_OFF+ (size_t)128*DD*4;         // 16 KB f32
static constexpr size_t ATTNT_OFF  = VMEAN_OFF+ (size_t)BB*DD*4;          // 6.3 MB f32 [B][L][48]
static constexpr size_t PART_OFF   = ATTNT_OFF+ (size_t)BB*LL*48*4;       // 6.3 MB f32 [B][8][48][512]

// ---------------------------------------------------------------------------
// Phase 0: split K (fp32) into f16 hi + f16 lo residual
// ---------------------------------------------------------------------------
struct alignas(8) H4 { _Float16 h[4]; };

__global__ __launch_bounds__(256) void split_f16_kernel(
    const float* __restrict__ src, _Float16* __restrict__ hi,
    _Float16* __restrict__ lo, int n4)
{
  int i4 = blockIdx.x * 256 + threadIdx.x;
  if (i4 >= n4) return;
  int i = i4 * 4;
  float4 x = *(const float4*)(src + i);
  float f[4] = {x.x, x.y, x.z, x.w};
  H4 a, c;
  #pragma unroll
  for (int j = 0; j < 4; ++j) {
    _Float16 hv = (_Float16)f[j];
    a.h[j] = hv;
    c.h[j] = (_Float16)(f[j] - (float)hv);
  }
  *(H4*)(hi + i) = a;
  *(H4*)(lo + i) = c;
}

// ---------------------------------------------------------------------------
// Phase 1: M[b,q] = (max_k - mean_k of raw q.k) * scale via f16x2 WMMA
//   grid: B * (L/QROWS) = 256 blocks of 256 threads (8 waves)
//   wave w owns rows [w*16, w*16+16); iterates all 4096 keys.
//   48 WMMAs per key tile round-robined over 4 accumulators so the
//   WMMA->WMMA RAW distance is 4 (hazard window is 1 NOP + 4 co-exec).
// ---------------------------------------------------------------------------
__global__ __launch_bounds__(256) void probsparse_m_kernel(
    const float*    __restrict__ q,
    const _Float16* __restrict__ k_hi,
    const _Float16* __restrict__ k_lo,
    float*          __restrict__ Mout)
{
  // ql stored fragment-ordered: [row][chunk][half][16 halves]  = 128 KB
  __shared__ __align__(16) _Float16 s_ql[QROWS * DD];
  __shared__ __align__(16) _Float16 s_kh[KTILE * DD];   // 16 KB
  __shared__ __align__(16) _Float16 s_kl[KTILE * DD];   // 16 KB

  const int b    = blockIdx.x >> 5;        // / (L/QROWS)=32
  const int qt   = blockIdx.x & 31;
  const int q0   = qt * QROWS;
  const int tid  = threadIdx.x;
  const int wave = tid >> 5;
  const int lane = tid & 31;
  const int lrow = lane & 15;
  const int lhalf= lane >> 4;

  const int myrow = wave * 16 + lrow;                       // [0,128)
  const size_t qbase = ((size_t)b * LL + q0 + myrow) * DD;  // fp32 Q row

  // ---- build qh fragments in registers, ql fragments in LDS ----
  // A layout (contiguous K, matches B): lane-half h holds K = d0 + h*16 .. +16
  v16h qh[16];
  #pragma unroll
  for (int c = 0; c < 16; ++c) {
    const int dA0 = c * 32 + lhalf * 16;
    const int dA1 = c * 32 + lhalf * 16 + 8;
    float f[16];
    {
      const float4* s0 = (const float4*)(q + qbase + dA0);
      const float4* s1 = (const float4*)(q + qbase + dA1);
      float4 t0 = s0[0], t1 = s0[1], t2 = s1[0], t3 = s1[1];
      f[0]=t0.x; f[1]=t0.y; f[2]=t0.z; f[3]=t0.w;
      f[4]=t1.x; f[5]=t1.y; f[6]=t1.z; f[7]=t1.w;
      f[8]=t2.x; f[9]=t2.y; f[10]=t2.z; f[11]=t2.w;
      f[12]=t3.x; f[13]=t3.y; f[14]=t3.z; f[15]=t3.w;
    }
    v16h h, l;
    #pragma unroll
    for (int e = 0; e < 16; ++e) {
      _Float16 hv = (_Float16)f[e];
      h[e] = hv;
      l[e] = (_Float16)(f[e] - (float)hv);
    }
    qh[c] = h;
    // fragment-ordered ql slot, private to this lane
    const int fi = (myrow * 16 + c) * 2 + lhalf;            // *16 halves
    *(v16h*)(&s_ql[fi * 16]) = l;
  }

  v8f m8, s8;
  #pragma unroll
  for (int vv = 0; vv < 8; ++vv) { m8[vv] = -3.402823466e+38f; s8[vv] = 0.0f; }

  const size_t kbase = (size_t)b * LL * DD;
  __syncthreads();

  for (int kt = 0; kt < NKT; ++kt) {
    // ---- stage 16-key tile of k_hi / k_lo into LDS (16 KB each) ----
    {
      const float4* sh = (const float4*)(k_hi + kbase + (size_t)kt * KTILE * DD);
      const float4* sl = (const float4*)(k_lo + kbase + (size_t)kt * KTILE * DD);
      float4* dh = (float4*)s_kh;
      float4* dl = (float4*)s_kl;
      #pragma unroll
      for (int j = 0; j < 4; ++j) {
        dh[tid + 256 * j] = sh[tid + 256 * j];
        dl[tid + 256 * j] = sl[tid + 256 * j];
      }
      if (kt + 1 < NKT) {
        __builtin_prefetch(k_hi + kbase + (size_t)(kt + 1) * KTILE * DD + tid * 32, 0, 1);
        __builtin_prefetch(k_lo + kbase + (size_t)(kt + 1) * KTILE * DD + tid * 32, 0, 1);
      }
    }
    __syncthreads();

    // 4 independent accumulators, round-robin across the 48 WMMAs
    v8f a4[4];
    #pragma unroll
    for (int r = 0; r < 4; ++r)
      #pragma unroll
      for (int vv = 0; vv < 8; ++vv) a4[r][vv] = 0.0f;

    // ---- accumulate 16x16 score tile over D: 3 WMMA per 32-wide chunk ----
    // B k order chosen here (same as A, not the ISA table order): lane = column(key), lane-half selects
    // contiguous K 0-15 / 16-31 -> one 32B contiguous LDS read per lane.
    #pragma unroll
    for (int c = 0; c < 16; ++c) {
      const int d0   = c * 32 + lhalf * 16;
      const int boff = lrow * DD + d0;
      v16h Bh = *(const v16h*)(&s_kh[boff]);
      v16h Bl = *(const v16h*)(&s_kl[boff]);
      const int fi = (myrow * 16 + c) * 2 + lhalf;
      v16h Al = *(const v16h*)(&s_ql[fi * 16]);
      const int r0 = (3 * c + 0) & 3;
      const int r1 = (3 * c + 1) & 3;
      const int r2 = (3 * c + 2) & 3;
      a4[r0] = __builtin_amdgcn_wmma_f32_16x16x32_f16(false, qh[c], false, Bh,
                                                      (short)0, a4[r0], false, false);
      a4[r1] = __builtin_amdgcn_wmma_f32_16x16x32_f16(false, qh[c], false, Bl,
                                                      (short)0, a4[r1], false, false);
      a4[r2] = __builtin_amdgcn_wmma_f32_16x16x32_f16(false, Al,   false, Bh,
                                                      (short)0, a4[r2], false, false);
    }

    #pragma unroll
    for (int vv = 0; vv < 8; ++vv) {
      float s = (a4[0][vv] + a4[1][vv]) + (a4[2][vv] + a4[3][vv]);
      m8[vv] = fmaxf(m8[vv], s);
      s8[vv] += s;
    }
    __syncthreads();   // protect tile overwrite next iteration
  }

  // ---- reduce across the 16 column lanes of each half (butterfly) ----
  #pragma unroll
  for (int msk = 1; msk <= 8; msk <<= 1) {
    #pragma unroll
    for (int vv = 0; vv < 8; ++vv) {
      m8[vv] = fmaxf(m8[vv], __shfl_xor(m8[vv], msk, 32));
      s8[vv] += __shfl_xor(s8[vv], msk, 32);
    }
  }
  // C/D layout: VGPR v, lane-half h -> row = v + 8*h; lanes 0 and 16 write.
  if (lrow == 0) {
    #pragma unroll
    for (int vv = 0; vv < 8; ++vv) {
      int row = wave * 16 + lhalf * 8 + vv;
      Mout[(size_t)b * LL + q0 + row] = (m8[vv] - s8[vv] * (1.0f / (float)LL)) * SCALE;
    }
  }
}

// ---------------------------------------------------------------------------
// Phase 2: top-41 per batch (iterative argmax, ties -> lowest index)
// ---------------------------------------------------------------------------
__global__ __launch_bounds__(256) void topk_kernel(
    const float* __restrict__ M, int* __restrict__ topk)
{
  __shared__ float vals[LL];
  __shared__ float rv[256];
  __shared__ int   ri[256];
  const int b = blockIdx.x, t = threadIdx.x;
  for (int i = t; i < LL; i += 256) vals[i] = M[(size_t)b * LL + i];
  __syncthreads();
  for (int it = 0; it < UU; ++it) {
    float bv = -3.402823466e+38f; int bi = LL;
    for (int i = t; i < LL; i += 256)
      if (vals[i] > bv || (vals[i] == bv && i < bi)) { bv = vals[i]; bi = i; }
    rv[t] = bv; ri[t] = bi;
    __syncthreads();
    for (int s = 128; s > 0; s >>= 1) {
      if (t < s) {
        if (rv[t + s] > rv[t] || (rv[t + s] == rv[t] && ri[t + s] < ri[t])) {
          rv[t] = rv[t + s]; ri[t] = ri[t + s];
        }
      }
      __syncthreads();
    }
    if (t == 0) { topk[b * 48 + it] = ri[0]; vals[ri[0]] = -3.402823466e+38f; }
    __syncthreads();
  }
}

// ---------------------------------------------------------------------------
// Phase 3: V column mean (two-stage, no atomics)
// ---------------------------------------------------------------------------
__global__ __launch_bounds__(256) void vmean_partial_kernel(
    const float* __restrict__ v, float* __restrict__ vpart)
{
  const int b = blockIdx.x >> 4, c = blockIdx.x & 15, t = threadIdx.x;
  #pragma unroll
  for (int j = 0; j < 2; ++j) {
    const int d = t + j * 256;
    float s = 0.0f;
    for (int r = 0; r < 256; ++r)
      s += v[((size_t)b * LL + c * 256 + r) * DD + d];
    vpart[(size_t)blockIdx.x * DD + d] = s;
  }
}

__global__ __launch_bounds__(256) void vmean_final_kernel(
    const float* __restrict__ vpart, float* __restrict__ vmean)
{
  const int b = blockIdx.x, t = threadIdx.x;
  #pragma unroll
  for (int j = 0; j < 2; ++j) {
    const int d = t + j * 256;
    float s = 0.0f;
    #pragma unroll
    for (int c = 0; c < 16; ++c) s += vpart[(size_t)(b * 16 + c) * DD + d];
    vmean[(size_t)b * DD + d] = s * (1.0f / (float)LL);
  }
}

// ---------------------------------------------------------------------------
// Phase 4a: exact fp32 scores + softmax for selected rows -> attnT [B][L][48]
// ---------------------------------------------------------------------------
__global__ __launch_bounds__(256) void attn_kernel(
    const float* __restrict__ q, const float* __restrict__ k,
    const int* __restrict__ topk, float* __restrict__ attnT)
{
  __shared__ __align__(16) float qs[DD];
  __shared__ float sc[LL];
  __shared__ float red[256];
  const int b = blockIdx.x / UU, u = blockIdx.x % UU, t = threadIdx.x;
  const int qrow = topk[b * 48 + u];
  for (int i = t; i < DD; i += 256) qs[i] = q[((size_t)b * LL + qrow) * DD + i];
  __syncthreads();

  const float4* q4 = (const float4*)qs;
  for (int kk = t; kk < LL; kk += 256) {
    const float4* kr = (const float4*)(k + ((size_t)b * LL + kk) * DD);
    float s = 0.0f;
    #pragma unroll 8
    for (int j = 0; j < DD / 4; ++j) {
      float4 a = q4[j], bb = kr[j];
      s += a.x * bb.x + a.y * bb.y + a.z * bb.z + a.w * bb.w;
    }
    sc[kk] = s * SCALE;
  }
  __syncthreads();

  float lm = -3.402823466e+38f;
  for (int kk = t; kk < LL; kk += 256) lm = fmaxf(lm, sc[kk]);
  red[t] = lm; __syncthreads();
  for (int s = 128; s > 0; s >>= 1) { if (t < s) red[t] = fmaxf(red[t], red[t + s]); __syncthreads(); }
  const float mx = red[0];
  __syncthreads();

  float ls = 0.0f;
  for (int kk = t; kk < LL; kk += 256) { float e = __expf(sc[kk] - mx); sc[kk] = e; ls += e; }
  red[t] = ls; __syncthreads();
  for (int s = 128; s > 0; s >>= 1) { if (t < s) red[t] += red[t + s]; __syncthreads(); }
  const float inv = 1.0f / red[0];
  __syncthreads();

  for (int kk = t; kk < LL; kk += 256)
    attnT[((size_t)b * LL + kk) * 48 + u] = sc[kk] * inv;
}

// ---------------------------------------------------------------------------
// Phase 4b: partial attnT^T x V over key chunks  -> partial [B][8][48][512]
// ---------------------------------------------------------------------------
__global__ __launch_bounds__(128) void pv_partial_kernel(
    const float* __restrict__ v, const float* __restrict__ attnT,
    float* __restrict__ partial)
{
  const int b  = blockIdx.x >> 5;
  const int dc = (blockIdx.x >> 3) & 3;
  const int kc = blockIdx.x & 7;
  const int d  = dc * 128 + threadIdx.x;

  float acc[UU];
  #pragma unroll
  for (int u = 0; u < UU; ++u) acc[u] = 0.0f;

  for (int kk = kc * 512; kk < kc * 512 + 512; ++kk) {
    const float vv = v[((size_t)b * LL + kk) * DD + d];
    const float* a = attnT + ((size_t)b * LL + kk) * 48;
    #pragma unroll
    for (int u = 0; u < UU; ++u) acc[u] += a[u] * vv;
  }
  #pragma unroll
  for (int u = 0; u < UU; ++u)
    partial[((size_t)(b * 8 + kc) * 48 + u) * DD + d] = acc[u];
}

// ---------------------------------------------------------------------------
// Phase 5: broadcast v_mean to all output rows
// ---------------------------------------------------------------------------
__global__ __launch_bounds__(256) void bcast_kernel(
    const float* __restrict__ vmean, float* __restrict__ out)
{
  const size_t i = ((size_t)blockIdx.x * 256 + threadIdx.x) * 4;
  const int b = (int)(i / ((size_t)LL * DD));
  const int d = (int)(i % DD);
  float4 vm = *(const float4*)(vmean + (size_t)b * DD + d);
  *(float4*)(out + i) = vm;
}

// ---------------------------------------------------------------------------
// Phase 4c: reduce key-chunk partials, scatter selected rows into out
// ---------------------------------------------------------------------------
__global__ __launch_bounds__(256) void scatter_kernel(
    const float* __restrict__ partial, const int* __restrict__ topk,
    float* __restrict__ out)
{
  const int b = blockIdx.x / UU, u = blockIdx.x % UU, t = threadIdx.x;
  const int row = topk[b * 48 + u];
  #pragma unroll
  for (int j = 0; j < 2; ++j) {
    const int d = t + j * 256;
    float s = 0.0f;
    #pragma unroll
    for (int kc = 0; kc < 8; ++kc)
      s += partial[((size_t)(b * 8 + kc) * 48 + u) * DD + d];
    out[((size_t)b * LL + row) * DD + d] = s;
  }
}

// ---------------------------------------------------------------------------
extern "C" void kernel_launch(void* const* d_in, const int* in_sizes, int n_in,
                              void* d_out, int out_size, void* d_ws, size_t ws_size,
                              hipStream_t stream) {
  const float* q = (const float*)d_in[0];
  const float* k = (const float*)d_in[1];
  const float* v = (const float*)d_in[2];
  float* out = (float*)d_out;
  char*  ws  = (char*)d_ws;

  _Float16* k_hi  = (_Float16*)(ws + K_HI_OFF);
  _Float16* k_lo  = (_Float16*)(ws + K_LO_OFF);
  float*    M     = (float*)(ws + M_OFF);
  int*      tk    = (int*)(ws + TOPK_OFF);
  float*    vpart = (float*)(ws + VPART_OFF);
  float*    vmean = (float*)(ws + VMEAN_OFF);
  float*    attnT = (float*)(ws + ATTNT_OFF);
  float*    part  = (float*)(ws + PART_OFF);

  const int n  = BB * LL * DD;          // 16,777,216
  const int n4 = n / 4;

  split_f16_kernel   <<<n4 / 256, 256, 0, stream>>>(k, k_hi, k_lo, n4);
  probsparse_m_kernel<<<BB * (LL / QROWS), 256, 0, stream>>>(q, k_hi, k_lo, M);
  topk_kernel        <<<BB, 256, 0, stream>>>(M, tk);
  vmean_partial_kernel<<<BB * 16, 256, 0, stream>>>(v, vpart);
  vmean_final_kernel <<<BB, 256, 0, stream>>>(vpart, vmean);
  attn_kernel        <<<BB * UU, 256, 0, stream>>>(q, k, tk, attnT);
  pv_partial_kernel  <<<BB * 32, 128, 0, stream>>>(v, attnT, part);
  bcast_kernel       <<<n4 / 256, 256, 0, stream>>>(vmean, out);
  scatter_kernel     <<<BB * UU, 256, 0, stream>>>(part, tk, out);
}
